// Mamba2Mixer_71004399337817
// MI455X (gfx1250) — matrix operand order repaired and checked by reading, not compiled, not run on hardware
//
#include <hip/hip_runtime.h>
#include <math.h>

// ---------------- problem constants ----------------
#define D_MODEL   2048
#define D_STATE   128
#define D_CONV    4
#define NHEADS    64
#define NGROUPS   8
#define HEAD_DIM  64
#define CHUNK     256
#define BATCH     2
#define SEQLEN    2048
#define T_TOK     (BATCH * SEQLEN)                 // 4096
#define D_INNER   (NHEADS * HEAD_DIM)              // 4096
#define CONV_DIM  (D_INNER + 2 * NGROUPS * D_STATE)// 6144
#define D_IN_PROJ (2*D_INNER + 2*NGROUPS*D_STATE + NHEADS) // 10304
#define NC        (SEQLEN / CHUNK)                 // 8
#define EPS_F     1e-5f

typedef __bf16 bf16;
typedef __attribute__((ext_vector_type(16))) __bf16 v16bf;
typedef __attribute__((ext_vector_type(8)))  float  v8f;

union Frag { v16bf v; unsigned u[8]; };

static __device__ __forceinline__ bf16 f2bf(float f) {
  unsigned u = __builtin_bit_cast(unsigned, f);
  unsigned short h = (unsigned short)((u + 0x7FFFu + ((u >> 16) & 1u)) >> 16); // RNE
  return __builtin_bit_cast(bf16, h);
}
static __device__ __forceinline__ unsigned pack2(float a, float b) {
  unsigned x = (unsigned)__builtin_bit_cast(unsigned short, f2bf(a));
  unsigned y = (unsigned)__builtin_bit_cast(unsigned short, f2bf(b));
  return x | (y << 16);
}
static __device__ __forceinline__ float siluf(float x) { return x / (1.f + __expf(-x)); }

// ---------------- K1: fp32 -> bf16 cast (vectorized x4) ----------------
__global__ void cast_f32_bf16(const float* __restrict__ src, bf16* __restrict__ dst, int n4) {
  int i = blockIdx.x * blockDim.x + threadIdx.x;
  if (i < n4) {
    float4 v = ((const float4*)src)[i];
    uint2 o;
    o.x = pack2(v.x, v.y);
    o.y = pack2(v.z, v.w);
    ((uint2*)dst)[i] = o;
  }
}

// ---------------- K2: C[M,N](f32) = A[M,K](bf16) @ B[N,K]^T(bf16) ----------------
// block = 256 threads (8 waves); tile 256(M) x 64(N); K step 32.
// Double-buffered async global->LDS (b128 per lane), s_wait_asynccnt pipelining.
// M%256==0, N%64==0, K%32==0 required (holds for all calls here).
__global__ __launch_bounds__(256) void gemm_bf16_tn(
    const bf16* __restrict__ A, const bf16* __restrict__ B, float* __restrict__ C,
    int M, int N, int K) {
  __shared__ bf16 sA[2][256 * 32];
  __shared__ bf16 sB[2][64 * 32];
  const int tid = threadIdx.x;
  const int w = tid >> 5, lane = tid & 31;
  const int r = lane & 15, half = lane >> 4;
  const int wm = w >> 1, wn = w & 1;
  const int m0 = blockIdx.y * 256, n0 = blockIdx.x * 64;
  v8f acc[4][2] = {};

  // async-copy one K-stage (32 columns) into buffer st&1 : 5 b128 issues/lane
  auto issue_stage = [&](int st) {
    const int bi = st & 1, kb = st << 5;
    unsigned baseA = (unsigned)(uintptr_t)(&sA[bi][0]);
    unsigned baseB = (unsigned)(uintptr_t)(&sB[bi][0]);
#pragma unroll
    for (int i = 0; i < 4; ++i) {                 // sA: 1024 chunks of 16B
      int vi = i * 256 + tid;
      unsigned long long gp = (unsigned long long)(uintptr_t)(
          A + (size_t)(m0 + (vi >> 2)) * K + kb + ((vi & 3) << 3));
      unsigned lo = baseA + ((unsigned)vi << 4);
      asm volatile("global_load_async_to_lds_b128 %0, %1, off"
                   :: "v"(lo), "v"(gp) : "memory");
    }
    {                                             // sB: 256 chunks of 16B
      int vi = tid;
      unsigned long long gp = (unsigned long long)(uintptr_t)(
          B + (size_t)(n0 + (vi >> 2)) * K + kb + ((vi & 3) << 3));
      unsigned lo = baseB + ((unsigned)vi << 4);
      asm volatile("global_load_async_to_lds_b128 %0, %1, off"
                   :: "v"(lo), "v"(gp) : "memory");
    }
  };

  const int NST = K >> 5;
  issue_stage(0);
  for (int st = 0; st < NST; ++st) {
    if (st + 1 < NST) {
      issue_stage(st + 1);                        // prefetch next stage (other buffer)
      asm volatile("s_wait_asynccnt 5" ::: "memory");  // stage st done, st+1 in flight
    } else {
      asm volatile("s_wait_asynccnt 0" ::: "memory");
    }
    __syncthreads();
    const unsigned* sAu = (const unsigned*)sA[st & 1];
    const unsigned* sBu = (const unsigned*)sB[st & 1];
    Frag bfr[2];
#pragma unroll
    for (int nt = 0; nt < 2; ++nt) {
      int nn = wn * 32 + nt * 16 + r;
#pragma unroll
      for (int v = 0; v < 8; ++v) bfr[nt].u[v] = sBu[nn * 16 + (v < 4 ? 0 : 8) + half * 4 + (v & 3)];
    }
#pragma unroll
    for (int mt = 0; mt < 4; ++mt) {
      Frag af;
      int rr = wm * 64 + mt * 16 + r;
#pragma unroll
      for (int v = 0; v < 8; ++v)
        af.u[v] = sAu[rr * 16 + (v < 4 ? 0 : 8) + half * 4 + (v & 3)];
#pragma unroll
      for (int nt = 0; nt < 2; ++nt)
        acc[mt][nt] = __builtin_amdgcn_wmma_f32_16x16x32_bf16(
            false, af.v, false, bfr[nt].v, (short)0, acc[mt][nt], false, false);
    }
    __syncthreads();                              // reads done before stage st+2 overwrites
  }
#pragma unroll
  for (int mt = 0; mt < 4; ++mt)
#pragma unroll
    for (int nt = 0; nt < 2; ++nt)
#pragma unroll
      for (int j = 0; j < 8; ++j) {
        int row = m0 + wm * 64 + mt * 16 + half * 8 + j;
        int col = n0 + wn * 32 + nt * 16 + r;
        C[(size_t)row * N + col] = acc[mt][nt][j];
      }
}

// ---------------- K3: depthwise causal conv(4) + SiLU ----------------
__global__ void conv_silu(const float* __restrict__ zx, const float* __restrict__ Wc,
                          const float* __restrict__ bc, float* __restrict__ xbc) {
  long idx = (long)blockIdx.x * blockDim.x + threadIdx.x;
  if (idx >= (long)T_TOK * CONV_DIM) return;
  int ch = (int)(idx % CONV_DIM);
  int t = (int)(idx / CONV_DIM);
  int s = t & (SEQLEN - 1);
  float acc = bc[ch];
#pragma unroll
  for (int k = 0; k < D_CONV; ++k) {
    int sp = s - (D_CONV - 1) + k;
    if (sp >= 0)
      acc += zx[(size_t)(t - s + sp) * D_IN_PROJ + D_INNER + ch] * Wc[ch * D_CONV + k];
  }
  xbc[idx] = siluf(acc);
}

// ---------------- K3b: dt = softplus(dt_raw + dt_bias) ----------------
__global__ void dt_softplus(const float* __restrict__ zx, const float* __restrict__ dtb,
                            float* __restrict__ dt) {
  int idx = blockIdx.x * blockDim.x + threadIdx.x;
  if (idx >= T_TOK * NHEADS) return;
  int h = idx & 63, t = idx >> 6;
  float v = zx[(size_t)t * D_IN_PROJ + (D_INNER + CONV_DIM) + h] + dtb[h];
  dt[idx] = (v > 20.f) ? v : logf(1.f + __expf(v));
}

// ---------------- K4: per-(b,c,h) cumsum of dt*A ----------------
__global__ void acum_kernel(const float* __restrict__ dt, const float* __restrict__ A,
                            float* __restrict__ acum) {
  int idx = blockIdx.x * blockDim.x + threadIdx.x;     // (b*NC+c)*NHEADS + h
  if (idx >= BATCH * NC * NHEADS) return;
  int h = idx & 63;
  int t0 = (idx >> 6) * CHUNK;
  float Ah = A[h], run = 0.f;
  float* out = acum + (size_t)idx * CHUNK;
  for (int s = 0; s < CHUNK; ++s) {
    run += dt[(size_t)(t0 + s) * NHEADS + h] * Ah;
    out[s] = run;
  }
}

// ---------------- K5: states[h,p,n] = sum_s decay(s)*dt(s)*x(s,p)*B(s,n) ----------------
__global__ __launch_bounds__(256) void states_kernel(
    const float* __restrict__ xbc, const float* __restrict__ dt,
    const float* __restrict__ acum, float* __restrict__ states) {
  __shared__ bf16 sX[64 * 32];   // A operand  [p][ss]
  __shared__ bf16 sB[128 * 32];  // B operand  [n][ss]
  const int bch = blockIdx.x;
  const int h = bch & 63, g = h >> 3;
  const int t0 = (bch >> 6) * CHUNK;
  const float* ac = acum + (size_t)bch * CHUNK;
  const float aclast = ac[CHUNK - 1];
  const int tid = threadIdx.x;
  const int w = tid >> 5, lane = tid & 31, r = lane & 15, half = lane >> 4;
  const int mt = w >> 1, ng = w & 1;
  v8f acc[4] = {};
  const unsigned* sXu = (const unsigned*)sX;
  const unsigned* sBu = (const unsigned*)sB;
  for (int kb = 0; kb < CHUNK; kb += 32) {
#pragma unroll
    for (int i = 0; i < 2; ++i) {                 // sX: transpose-stage, b128 reads
      int vi = i * 256 + tid;
      int ss = vi >> 4, p = (vi & 15) << 2;
      int s = kb + ss;
      float wgt = __expf(aclast - ac[s]) * dt[(size_t)(t0 + s) * NHEADS + h];
      float4 xv = *(const float4*)(xbc + (size_t)(t0 + s) * CONV_DIM + h * HEAD_DIM + p);
      sX[(p + 0) * 32 + ss] = f2bf(xv.x * wgt);
      sX[(p + 1) * 32 + ss] = f2bf(xv.y * wgt);
      sX[(p + 2) * 32 + ss] = f2bf(xv.z * wgt);
      sX[(p + 3) * 32 + ss] = f2bf(xv.w * wgt);
    }
#pragma unroll
    for (int i = 0; i < 4; ++i) {                 // sB: transpose-stage, b128 reads
      int vi = i * 256 + tid;
      int ss = vi >> 5, n = (vi & 31) << 2;
      float4 bv = *(const float4*)(xbc + (size_t)(t0 + kb + ss) * CONV_DIM + D_INNER + g * D_STATE + n);
      sB[(n + 0) * 32 + ss] = f2bf(bv.x);
      sB[(n + 1) * 32 + ss] = f2bf(bv.y);
      sB[(n + 2) * 32 + ss] = f2bf(bv.z);
      sB[(n + 3) * 32 + ss] = f2bf(bv.w);
    }
    __syncthreads();
    Frag af;
    int pr = mt * 16 + r;
#pragma unroll
    for (int v = 0; v < 8; ++v)
      af.u[v] = sXu[pr * 16 + (v < 4 ? 0 : 8) + half * 4 + (v & 3)];
#pragma unroll
    for (int i = 0; i < 4; ++i) {
      Frag bfr;
      int nn = (ng * 4 + i) * 16 + r;
#pragma unroll
      for (int v = 0; v < 8; ++v) bfr.u[v] = sBu[nn * 16 + (v < 4 ? 0 : 8) + half * 4 + (v & 3)];
      acc[i] = __builtin_amdgcn_wmma_f32_16x16x32_bf16(
          false, af.v, false, bfr.v, (short)0, acc[i], false, false);
    }
    __syncthreads();
  }
  float* out = states + (size_t)bch * HEAD_DIM * D_STATE;
#pragma unroll
  for (int i = 0; i < 4; ++i)
#pragma unroll
    for (int j = 0; j < 8; ++j) {
      int p = mt * 16 + half * 8 + j;
      int n = (ng * 4 + i) * 16 + r;
      out[p * D_STATE + n] = acc[i][j];
    }
}

// ---------------- K6: inter-chunk scan (8 steps, serial per element) ----------------
__global__ void scan_kernel(const float* __restrict__ states, const float* __restrict__ acum,
                            float* __restrict__ prev) {
  const int PER = HEAD_DIM * D_STATE;  // 8192
  int idx = blockIdx.x * blockDim.x + threadIdx.x;
  if (idx >= BATCH * NHEADS * PER) return;
  int e = idx % PER;
  int h = (idx / PER) & 63;
  int b = idx / (PER * NHEADS);
  float run = 0.f;
  for (int c = 0; c < NC; ++c) {
    int bch = (b * NC + c) * NHEADS + h;
    size_t base = (size_t)bch * PER + e;
    prev[base] = run;
    float dec = __expf(acum[(size_t)bch * CHUNK + CHUNK - 1]);
    run = run * dec + states[base];
  }
}

// ---------------- K7: Y = Y_diag + Y_off + D*x  (3 chained WMMA GEMMs) ----------------
__global__ __launch_bounds__(256) void y_kernel(
    const float* __restrict__ xbc, const float* __restrict__ dt,
    const float* __restrict__ acum, const float* __restrict__ prev,
    const float* __restrict__ Dp, float* __restrict__ y) {
  __shared__ bf16 sS1[64 * CHUNK];    // masked/scaled C·B^T, [l][s]  (32KB)
  __shared__ bf16 sStage[CHUNK * 32]; // reused: B[s][nn] / x[p][ss] / prev[p][nn]
  __shared__ bf16 sA[64 * 32];        // C tile staging [ll][nn]
  __shared__ float sAc[CHUNK];
  __shared__ float sDt[CHUNK];
  const int blk = blockIdx.x;
  const int lt = blk & 3, bch = blk >> 2;
  const int h = bch & 63, g = h >> 3;
  const int t0 = (bch >> 6) * CHUNK;
  const int l0 = lt * 64;
  const int tid = threadIdx.x;
  const int w = tid >> 5, lane = tid & 31, r = lane & 15, half = lane >> 4;
  const float* ac = acum + (size_t)bch * CHUNK;
  const unsigned* sS1u = (const unsigned*)sS1;
  const unsigned* sStu = (const unsigned*)sStage;
  const unsigned* sAu = (const unsigned*)sA;
  const size_t C_OFF = (size_t)(D_INNER + NGROUPS * D_STATE + g * D_STATE);
  const size_t B_OFF = (size_t)(D_INNER + g * D_STATE);

  for (int e = tid; e < CHUNK; e += 256) {
    sAc[e] = ac[e];
    sDt[e] = dt[(size_t)(t0 + e) * NHEADS + h];
  }
  __syncthreads();

  // ---- GEMM1: G[l,s] = sum_n C[l,n] B[s,n]   (M=64 l, N=256 s, K=128 n) ----
  {
    const int lw = w >> 1, sg = w & 1;
    v8f acc1[8] = {};
    for (int kb = 0; kb < D_STATE; kb += 32) {
#pragma unroll
      for (int i = 0; i < 2; ++i) {               // sA: direct-stage [ll][nn]
        int vi = i * 256 + tid;
        int ll = vi >> 3, nn = (vi & 7) << 2;
        float4 cv = *(const float4*)(xbc + (size_t)(t0 + l0 + ll) * CONV_DIM + C_OFF + kb + nn);
        uint2 o; o.x = pack2(cv.x, cv.y); o.y = pack2(cv.z, cv.w);
        ((uint2*)sA)[vi] = o;
      }
#pragma unroll
      for (int i = 0; i < 8; ++i) {               // sStage: direct-stage [s][nn]
        int vi = i * 256 + tid;
        int s = vi >> 3, nn = (vi & 7) << 2;
        float4 bv = *(const float4*)(xbc + (size_t)(t0 + s) * CONV_DIM + B_OFF + kb + nn);
        uint2 o; o.x = pack2(bv.x, bv.y); o.y = pack2(bv.z, bv.w);
        ((uint2*)sStage)[vi] = o;
      }
      __syncthreads();
      Frag af;
      int ll = lw * 16 + r;
#pragma unroll
      for (int v = 0; v < 8; ++v)
        af.u[v] = sAu[ll * 16 + (v < 4 ? 0 : 8) + half * 4 + (v & 3)];
#pragma unroll
      for (int i = 0; i < 8; ++i) {
        Frag bfr;
        int sc = sg * 128 + i * 16 + r;
#pragma unroll
        for (int v = 0; v < 8; ++v) bfr.u[v] = sStu[sc * 16 + (v < 4 ? 0 : 8) + half * 4 + (v & 3)];
        acc1[i] = __builtin_amdgcn_wmma_f32_16x16x32_bf16(
            false, af.v, false, bfr.v, (short)0, acc1[i], false, false);
      }
      __syncthreads();
    }
    // mask with L, fold exp(seg)*dt, write bf16 S1
#pragma unroll
    for (int i = 0; i < 8; ++i)
#pragma unroll
      for (int j = 0; j < 8; ++j) {
        int ll = lw * 16 + half * 8 + j;
        int lc = l0 + ll;
        int s = sg * 128 + i * 16 + r;
        float v = (s <= lc) ? acc1[i][j] * __expf(sAc[lc] - sAc[s]) * sDt[s] : 0.f;
        sS1[ll * CHUNK + s] = f2bf(v);
      }
  }
  __syncthreads();

  // ---- GEMM2: Yd[l,p] = sum_s S1[l,s] x[s,p]  and  GEMM-off: sum_n C[l,n] prev[p,n] ----
  const int lw2 = w >> 1, ph = w & 1;
  v8f acc2[2] = {};
  v8f accO[2] = {};
  for (int kb = 0; kb < CHUNK; kb += 32) {
#pragma unroll
    for (int i = 0; i < 2; ++i) {                 // x transpose-stage [p][ss]
      int vi = i * 256 + tid;
      int ss = vi >> 4, p = (vi & 15) << 2;
      float4 xv = *(const float4*)(xbc + (size_t)(t0 + kb + ss) * CONV_DIM + h * HEAD_DIM + p);
      sStage[(p + 0) * 32 + ss] = f2bf(xv.x);
      sStage[(p + 1) * 32 + ss] = f2bf(xv.y);
      sStage[(p + 2) * 32 + ss] = f2bf(xv.z);
      sStage[(p + 3) * 32 + ss] = f2bf(xv.w);
    }
    __syncthreads();
    Frag af;
    int ll = lw2 * 16 + r;
#pragma unroll
    for (int v = 0; v < 8; ++v)
      af.u[v] = sS1u[ll * (CHUNK / 2) + (kb >> 1) + (v < 4 ? 0 : 8) + half * 4 + (v & 3)];
#pragma unroll
    for (int i = 0; i < 2; ++i) {
      Frag bfr;
      int p = (ph * 2 + i) * 16 + r;
#pragma unroll
      for (int v = 0; v < 8; ++v) bfr.u[v] = sStu[p * 16 + (v < 4 ? 0 : 8) + half * 4 + (v & 3)];
      acc2[i] = __builtin_amdgcn_wmma_f32_16x16x32_bf16(
          false, af.v, false, bfr.v, (short)0, acc2[i], false, false);
    }
    __syncthreads();
  }
  const float* prevh = prev + (size_t)bch * HEAD_DIM * D_STATE;
  for (int kb = 0; kb < D_STATE; kb += 32) {
#pragma unroll
    for (int i = 0; i < 2; ++i) {                 // sA: C tile [ll][nn]
      int vi = i * 256 + tid;
      int ll = vi >> 3, nn = (vi & 7) << 2;
      float4 cv = *(const float4*)(xbc + (size_t)(t0 + l0 + ll) * CONV_DIM + C_OFF + kb + nn);
      uint2 o; o.x = pack2(cv.x, cv.y); o.y = pack2(cv.z, cv.w);
      ((uint2*)sA)[vi] = o;
    }
#pragma unroll
    for (int i = 0; i < 2; ++i) {                 // prev direct-stage [p][nn]
      int vi = i * 256 + tid;
      int p = vi >> 3, nn = (vi & 7) << 2;
      float4 pv = *(const float4*)(prevh + p * D_STATE + kb + nn);
      uint2 o; o.x = pack2(pv.x, pv.y); o.y = pack2(pv.z, pv.w);
      ((uint2*)sStage)[vi] = o;
    }
    __syncthreads();
    Frag af;
    int ll = lw2 * 16 + r;
#pragma unroll
    for (int v = 0; v < 8; ++v)
      af.u[v] = sAu[ll * 16 + (v < 4 ? 0 : 8) + half * 4 + (v & 3)];
#pragma unroll
    for (int i = 0; i < 2; ++i) {
      Frag bfr;
      int p = (ph * 2 + i) * 16 + r;
#pragma unroll
      for (int v = 0; v < 8; ++v) bfr.u[v] = sStu[p * 16 + (v < 4 ? 0 : 8) + half * 4 + (v & 3)];
      accO[i] = __builtin_amdgcn_wmma_f32_16x16x32_bf16(
          false, af.v, false, bfr.v, (short)0, accO[i], false, false);
    }
    __syncthreads();
  }
  float Dh = Dp[h];
#pragma unroll
  for (int i = 0; i < 2; ++i)
#pragma unroll
    for (int j = 0; j < 8; ++j) {
      int ll = lw2 * 16 + half * 8 + j;
      int lc = l0 + ll;
      int p = (ph * 2 + i) * 16 + r;
      size_t t = (size_t)(t0 + lc);
      float xv = xbc[t * CONV_DIM + h * HEAD_DIM + p];
      float val = acc2[i][j] + __expf(sAc[lc]) * accO[i][j] + Dh * xv;
      y[t * D_INNER + h * HEAD_DIM + p] = val;
    }
}

// ---------------- K8: gate (silu(z)) + grouped RMSNorm -> bf16 ----------------
__global__ __launch_bounds__(256) void gate_norm(
    const float* __restrict__ zx, const float* __restrict__ y,
    const float* __restrict__ nw, bf16* __restrict__ out) {
  __shared__ float red[256];
  const int GSZ = D_INNER / NGROUPS;  // 512
  int blk = blockIdx.x;
  int gi = blk & 7, t = blk >> 3;
  int tid = threadIdx.x;
  int d0 = gi * GSZ + tid, d1 = d0 + 256;
  float z0 = zx[(size_t)t * D_IN_PROJ + d0];
  float z1 = zx[(size_t)t * D_IN_PROJ + d1];
  float g0 = y[(size_t)t * D_INNER + d0] * siluf(z0);
  float g1 = y[(size_t)t * D_INNER + d1] * siluf(z1);
  red[tid] = g0 * g0 + g1 * g1;
  __syncthreads();
  for (int off = 128; off > 0; off >>= 1) {
    if (tid < off) red[tid] += red[tid + off];
    __syncthreads();
  }
  float rinv = rsqrtf(red[0] / (float)GSZ + EPS_F);
  out[(size_t)t * D_INNER + d0] = f2bf(g0 * rinv * nw[d0]);
  out[(size_t)t * D_INNER + d1] = f2bf(g1 * rinv * nw[d1]);
}

// ---------------- workspace layout (bytes) ----------------
#define ALIGN_UP(x) (((x) + 255) & ~(size_t)255)
static const size_t OFF_HID_BF = 0;
static const size_t SZ_HID_BF = (size_t)T_TOK * D_MODEL * 2;
static const size_t OFF_WIN_BF = ALIGN_UP(OFF_HID_BF + SZ_HID_BF);
static const size_t SZ_WIN_BF = (size_t)D_IN_PROJ * D_MODEL * 2;
static const size_t OFF_WOUT_BF = ALIGN_UP(OFF_WIN_BF + SZ_WIN_BF);
static const size_t SZ_WOUT_BF = (size_t)D_MODEL * D_INNER * 2;
static const size_t OFF_ZX = ALIGN_UP(OFF_WOUT_BF + SZ_WOUT_BF);
static const size_t SZ_ZX = (size_t)T_TOK * D_IN_PROJ * 4;
static const size_t OFF_XBC = ALIGN_UP(OFF_ZX + SZ_ZX);
static const size_t SZ_XBC = (size_t)T_TOK * CONV_DIM * 4;
static const size_t OFF_DT = ALIGN_UP(OFF_XBC + SZ_XBC);
static const size_t SZ_DT = (size_t)T_TOK * NHEADS * 4;
static const size_t OFF_ACUM = ALIGN_UP(OFF_DT + SZ_DT);
static const size_t SZ_ACUM = (size_t)BATCH * NC * NHEADS * CHUNK * 4;
static const size_t OFF_STATES = ALIGN_UP(OFF_ACUM + SZ_ACUM);
static const size_t SZ_STATES = (size_t)BATCH * NC * NHEADS * HEAD_DIM * D_STATE * 4;
static const size_t OFF_PREV = ALIGN_UP(OFF_STATES + SZ_STATES);
static const size_t OFF_Y = ALIGN_UP(OFF_PREV + SZ_STATES);
static const size_t SZ_Y = (size_t)T_TOK * D_INNER * 4;
static const size_t OFF_NRM_BF = ALIGN_UP(OFF_Y + SZ_Y);

extern "C" void kernel_launch(void* const* d_in, const int* in_sizes, int n_in,
                              void* d_out, int out_size, void* d_ws, size_t ws_size,
                              hipStream_t stream) {
  (void)in_sizes; (void)n_in; (void)out_size; (void)ws_size;
  const float* hidden  = (const float*)d_in[0];
  const float* W_in    = (const float*)d_in[1];
  const float* W_conv  = (const float*)d_in[2];
  const float* b_conv  = (const float*)d_in[3];
  const float* A       = (const float*)d_in[4];
  const float* Dp      = (const float*)d_in[5];
  const float* dt_bias = (const float*)d_in[6];
  const float* norm_w  = (const float*)d_in[7];
  const float* W_out   = (const float*)d_in[8];
  char* ws = (char*)d_ws;
  bf16*  hid_bf  = (bf16*)(ws + OFF_HID_BF);
  bf16*  win_bf  = (bf16*)(ws + OFF_WIN_BF);
  bf16*  wout_bf = (bf16*)(ws + OFF_WOUT_BF);
  float* zx      = (float*)(ws + OFF_ZX);
  float* xbc     = (float*)(ws + OFF_XBC);
  float* dt      = (float*)(ws + OFF_DT);
  float* acum    = (float*)(ws + OFF_ACUM);
  float* states  = (float*)(ws + OFF_STATES);
  float* prev    = (float*)(ws + OFF_PREV);
  float* y       = (float*)(ws + OFF_Y);
  bf16*  nrm_bf  = (bf16*)(ws + OFF_NRM_BF);
  float* out     = (float*)d_out;

  // casts (all sizes divisible by 4)
  {
    int n4 = (T_TOK * D_MODEL) / 4;
    cast_f32_bf16<<<(n4 + 255) / 256, 256, 0, stream>>>(hidden, hid_bf, n4);
    n4 = (D_IN_PROJ * D_MODEL) / 4;
    cast_f32_bf16<<<(n4 + 255) / 256, 256, 0, stream>>>(W_in, win_bf, n4);
    n4 = (D_MODEL * D_INNER) / 4;
    cast_f32_bf16<<<(n4 + 255) / 256, 256, 0, stream>>>(W_out, wout_bf, n4);
  }
  // in-projection: zx = hidden @ W_in^T   (M=4096, N=10304, K=2048)
  gemm_bf16_tn<<<dim3(D_IN_PROJ / 64, T_TOK / 256), 256, 0, stream>>>(
      hid_bf, win_bf, zx, T_TOK, D_IN_PROJ, D_MODEL);
  // conv + silu, dt softplus
  {
    long n = (long)T_TOK * CONV_DIM;
    conv_silu<<<(unsigned)((n + 255) / 256), 256, 0, stream>>>(zx, W_conv, b_conv, xbc);
    int m = T_TOK * NHEADS;
    dt_softplus<<<(m + 255) / 256, 256, 0, stream>>>(zx, dt_bias, dt);
  }
  // cumsum a
  acum_kernel<<<(BATCH * NC * NHEADS + 255) / 256, 256, 0, stream>>>(dt, A, acum);
  // chunk states (WMMA)
  states_kernel<<<BATCH * NC * NHEADS, 256, 0, stream>>>(xbc, dt, acum, states);
  // inter-chunk scan
  {
    int n = BATCH * NHEADS * HEAD_DIM * D_STATE;
    scan_kernel<<<(n + 255) / 256, 256, 0, stream>>>(states, acum, prev);
  }
  // Y = Y_diag + Y_off + D*x (WMMA)
  y_kernel<<<BATCH * NC * NHEADS * 4, 256, 0, stream>>>(xbc, dt, acum, prev, Dp, y);
  // gated RMSNorm -> bf16
  gate_norm<<<T_TOK * NGROUPS, 256, 0, stream>>>(zx, y, norm_w, nrm_bf);
  // out-projection: out = normed @ W_out^T  (M=4096, N=2048, K=4096)
  gemm_bf16_tn<<<dim3(D_MODEL / 64, T_TOK / 256), 256, 0, stream>>>(
      nrm_bf, wout_bf, out, T_TOK, D_MODEL, D_INNER);
}
